// MHRetention_5085241278932
// MI455X (gfx1250) — matrix operand order repaired and checked by reading, not compiled, not run on hardware
//
#include <hip/hip_runtime.h>
#include <hip/hip_bf16.h>

typedef _Float16 v16h __attribute__((ext_vector_type(16)));
typedef _Float16 v8h  __attribute__((ext_vector_type(8)));
typedef _Float16 v4h  __attribute__((ext_vector_type(4)));
typedef float    v8f  __attribute__((ext_vector_type(8)));
typedef float    v4f  __attribute__((ext_vector_type(4)));

constexpr int EMB   = 2048;
constexpr int HEADS = 16;
constexpr int HD    = 128;
constexpr int BATCH = 4;
constexpr int SEQ   = 4096;
constexpr int MTOT  = BATCH * SEQ;                 // 16384 rows
constexpr float INV_SCALE = 0.08838834764831845f;  // 1/sqrt(128)
constexpr float SRMS      = 0.08838834764831845f;  // 1/sqrt(128)

// ---------------------------------------------------------------------------
// Fragment loaders (global, f16, K-contiguous rows).
// A 16x32 f16 layout: lanes 0-15 hold K=k0..k0+7 (+16..23), lanes 16-31 hold
// K=k0+8..15 (+24..31)  -> two 16B loads at +0 and +16 elements, base k0+8*hi.
// B 32x16 f16 layout: same K order as A -> two 16B loads at +0 and +16
// elements, base k0+8*hi.
// ---------------------------------------------------------------------------
__device__ __forceinline__ v16h frag_a(const _Float16* __restrict__ row0,
                                       int ld, int lane, int k0) {
  const _Float16* p = row0 + (size_t)(lane & 15) * ld + k0 + ((lane >> 4) << 3);
  v8h lo = *(const v8h*)p;
  v8h hi = *(const v8h*)(p + 16);
  v16h f;
#pragma unroll
  for (int i = 0; i < 8; ++i) { f[i] = lo[i]; f[i + 8] = hi[i]; }
  return f;
}

__device__ __forceinline__ v16h frag_b(const _Float16* __restrict__ row0,
                                       int ld, int lane, int k0) {
  const _Float16* p = row0 + (size_t)(lane & 15) * ld + k0 + ((lane >> 4) << 3);
  v8h lo = *(const v8h*)p;
  v8h hi = *(const v8h*)(p + 16);
  v16h f;
#pragma unroll
  for (int i = 0; i < 8; ++i) { f[i] = lo[i]; f[i + 8] = hi[i]; }
  return f;
}

// Per-wave 32(M) x 64(N) tile = 2x4 WMMA accumulators, K-loop step 32.
// Plain rolled form, unrolled 2x: both K-step batches in the body are
// independent (no loop-carried fragment values), so the scheduler can hoist
// the second batch's loads above the first batch's WMMAs with free register
// renaming -> no rotation movs, one latency exposure per 64 K.
__device__ __forceinline__ void wave_mma(const _Float16* __restrict__ A, int lda,
                                         const _Float16* __restrict__ B, int ldb,
                                         int K, int lane, v8f (&acc)[2][4]) {
#pragma unroll 2
  for (int k0 = 0; k0 < K; k0 += 32) {
    v16h a[2], b[4];
#pragma unroll
    for (int i = 0; i < 2; ++i) a[i] = frag_a(A + (size_t)(i * 16) * lda, lda, lane, k0);
#pragma unroll
    for (int j = 0; j < 4; ++j) b[j] = frag_b(B + (size_t)(j * 16) * ldb, ldb, lane, k0);
#pragma unroll
    for (int i = 0; i < 2; ++i)
#pragma unroll
      for (int j = 0; j < 4; ++j)
        acc[i][j] = __builtin_amdgcn_wmma_f32_16x16x32_f16(
            false, a[i], false, b[j], (short)0, acc[i][j], false, false);
  }
}

// ---------------------------------------------------------------------------
// Projection GEMM: C[M=16384, N=2048] = A[M,K] * W[N,K]^T
// MODE 0: relu(c)/sqrt(hd)  -> f16 natural (Q)
// MODE 1: relu(c)/sqrt(hd)  -> f16 transposed per-head (B,H,hd,L)  (Kt)
// MODE 2: c                 -> f16 transposed per-head            (Vt)
// MODE 3: silu(c)           -> f16 natural (U)
// MODE 4: c                 -> f32 natural (final output)
// ---------------------------------------------------------------------------
template <int MODE>
__global__ __launch_bounds__(256) void gemm_proj(const _Float16* __restrict__ A,
                                                 const _Float16* __restrict__ W,
                                                 void* __restrict__ Out, int K) {
  const int lane = threadIdx.x & 31;
  const int wave = threadIdx.x >> 5;
  const int wm = wave & 3, wn = wave >> 2;
  const int m0 = blockIdx.y * 128 + wm * 32;
  const int n0 = blockIdx.x * 128 + wn * 64;

  v8f acc[2][4] = {};
  wave_mma(A + (size_t)m0 * K, K, W + (size_t)n0 * K, K, K, lane, acc);

  const bool hi = lane >= 16;
  const int nl = lane & 15;
#pragma unroll
  for (int i = 0; i < 2; ++i) {
    const int mb = m0 + i * 16 + (hi ? 8 : 0);
#pragma unroll
    for (int j = 0; j < 4; ++j) {
      const int n = n0 + j * 16 + nl;
      if (MODE == 0 || MODE == 3) {
        _Float16* O = (_Float16*)Out;
#pragma unroll
        for (int r = 0; r < 8; ++r) {
          float v = acc[i][j][r];
          if (MODE == 0) v = fmaxf(v, 0.f) * INV_SCALE;
          else           v = v / (1.f + __expf(-v));
          O[(size_t)(mb + r) * EMB + n] = (_Float16)v;
        }
      } else if (MODE == 4) {
        float* O = (float*)Out;
#pragma unroll
        for (int r = 0; r < 8; ++r) O[(size_t)(mb + r) * EMB + n] = acc[i][j][r];
      } else {  // MODE 1 / 2 : transposed per-head store, contiguous along L
        _Float16* T = (_Float16*)Out;
        const int bb = mb / SEQ;
        const int l0 = mb % SEQ;
        const int h = n >> 7, d = n & (HD - 1);
        v8h pk;
#pragma unroll
        for (int r = 0; r < 8; ++r) {
          float v = acc[i][j][r];
          if (MODE == 1) v = fmaxf(v, 0.f) * INV_SCALE;
          pk[r] = (_Float16)v;
        }
        *(v8h*)(T + ((((size_t)bb * HEADS + h) * HD + d) * SEQ + l0)) = pk;
      }
    }
  }
}

// KVt[bh][e][d] = sum_l Vt[bh][e][l] * Kt[bh][d][l]   (128x128, K = 4096)
__global__ __launch_bounds__(256) void gemm_kv(const _Float16* __restrict__ Vt,
                                               const _Float16* __restrict__ Kt,
                                               _Float16* __restrict__ KVt) {
  const int lane = threadIdx.x & 31;
  const int wave = threadIdx.x >> 5;
  const int wm = wave & 3, wn = wave >> 2;
  const int bh = blockIdx.x;
  const int m0 = wm * 32, n0 = wn * 64;

  const _Float16* A = Vt + ((size_t)bh * HD + m0) * SEQ;
  const _Float16* B = Kt + ((size_t)bh * HD + n0) * SEQ;
  v8f acc[2][4] = {};
  wave_mma(A, SEQ, B, SEQ, SEQ, lane, acc);

  const bool hi = lane >= 16;
  const int nl = lane & 15;
  _Float16* O = KVt + (size_t)bh * HD * HD;
#pragma unroll
  for (int i = 0; i < 2; ++i) {
    const int mb = m0 + i * 16 + (hi ? 8 : 0);
#pragma unroll
    for (int j = 0; j < 4; ++j) {
      const int n = n0 + j * 16 + nl;
#pragma unroll
      for (int r = 0; r < 8; ++r)
        O[(size_t)(mb + r) * HD + n] = (_Float16)acc[i][j][r];
    }
  }
}

// O[b,l,h,e] = sum_d Q[b,l,h,d] * KVt[bh][e][d]   (M=4096 per bh, N=128, K=128)
__global__ __launch_bounds__(256) void gemm_o(const _Float16* __restrict__ Q,
                                              const _Float16* __restrict__ KVt,
                                              _Float16* __restrict__ O) {
  const int lane = threadIdx.x & 31;
  const int wave = threadIdx.x >> 5;
  const int wm = wave & 3, wn = wave >> 2;
  const int bh = blockIdx.y;
  const int b = bh >> 4, h = bh & 15;
  const int m0 = blockIdx.x * 128 + wm * 32;  // within SEQ
  const int n0 = wn * 64;

  const _Float16* A = Q + ((size_t)(b * SEQ + m0)) * EMB + h * HD;
  const _Float16* B = KVt + (size_t)bh * HD * HD + (size_t)n0 * HD;
  v8f acc[2][4] = {};
  wave_mma(A, EMB, B, HD, HD, lane, acc);

  const bool hi = lane >= 16;
  const int nl = lane & 15;
#pragma unroll
  for (int i = 0; i < 2; ++i) {
    const int mb = m0 + i * 16 + (hi ? 8 : 0);
#pragma unroll
    for (int j = 0; j < 4; ++j) {
      const int n = n0 + j * 16 + nl;
#pragma unroll
      for (int r = 0; r < 8; ++r)
        O[((size_t)(b * SEQ + mb + r)) * EMB + h * HD + n] = (_Float16)acc[i][j][r];
    }
  }
}

// SRMSNorm over each 128-wide head row, times precomputed silu(U). In place on O.
__global__ __launch_bounds__(256) void srms_mul(_Float16* __restrict__ O,
                                                const _Float16* __restrict__ U) {
  const int lane = threadIdx.x & 31;
  const int wave = threadIdx.x >> 5;
  const size_t seg = (size_t)blockIdx.x * 8 + wave;  // B*L*H segments
  const size_t m = seg >> 4;
  const int h = (int)(seg & 15);

  _Float16* op = O + m * EMB + h * HD + lane * 4;
  const _Float16* up = U + m * EMB + h * HD + lane * 4;
  v4h o = *(const v4h*)op;
  float s = 0.f;
#pragma unroll
  for (int r = 0; r < 4; ++r) { float f = (float)o[r]; s += f * f; }
#pragma unroll
  for (int off = 16; off > 0; off >>= 1) s += __shfl_xor(s, off, 32);
  const float norm = sqrtf(s) * SRMS;
  const float inv = 1.f / fmaxf(norm, 1e-12f);
  v4h u = *(const v4h*)up;
  v4h res;
#pragma unroll
  for (int r = 0; r < 4; ++r) res[r] = (_Float16)((float)o[r] * inv * (float)u[r]);
  *(v4h*)op = res;
}

// fp32 -> fp16 streaming convert (n4 = count of float4 groups)
__global__ __launch_bounds__(256) void cvt_f32_f16(const float* __restrict__ in,
                                                   _Float16* __restrict__ out,
                                                   size_t n4) {
  const size_t i = (size_t)blockIdx.x * blockDim.x + threadIdx.x;
  if (i < n4) {
    v4f v = *(const v4f*)(in + i * 4);
    v4h o;
#pragma unroll
    for (int r = 0; r < 4; ++r) o[r] = (_Float16)v[r];
    *(v4h*)(out + i * 4) = o;
  }
}

extern "C" void kernel_launch(void* const* d_in, const int* in_sizes, int n_in,
                              void* d_out, int out_size, void* d_ws, size_t ws_size,
                              hipStream_t stream) {
  const float* x  = (const float*)d_in[0];
  const float* wq = (const float*)d_in[1];
  const float* wk = (const float*)d_in[2];
  const float* wv = (const float*)d_in[3];
  const float* wu = (const float*)d_in[4];
  const float* wo = (const float*)d_in[5];

  char* ws = (char*)d_ws;
  size_t off = 0;
  auto alloc_h = [&](size_t elems) -> _Float16* {
    _Float16* p = (_Float16*)(ws + off);
    off += elems * sizeof(_Float16);
    off = (off + 255) & ~(size_t)255;
    return p;
  };

  const size_t NXW = (size_t)MTOT * EMB;   // 33.5M
  const size_t NW  = (size_t)EMB * EMB;    // 4.2M
  _Float16* xh  = alloc_h(NXW);
  _Float16* wqh = alloc_h(NW);
  _Float16* wkh = alloc_h(NW);
  _Float16* wvh = alloc_h(NW);
  _Float16* wuh = alloc_h(NW);
  _Float16* woh = alloc_h(NW);
  _Float16* Qb  = alloc_h(NXW);
  _Float16* Ktb = alloc_h(NXW);            // (B,H,hd,L)
  _Float16* Vtb = alloc_h(NXW);            // (B,H,hd,L)
  _Float16* Ub  = alloc_h(NXW);
  _Float16* KVb = alloc_h((size_t)BATCH * HEADS * HD * HD);
  _Float16* Ob  = alloc_h(NXW);
  (void)ws_size; (void)in_sizes; (void)n_in; (void)out_size;

  // 1) fp32 -> fp16
  cvt_f32_f16<<<(unsigned)((NXW / 4 + 255) / 256), 256, 0, stream>>>(x, xh, NXW / 4);
  cvt_f32_f16<<<(unsigned)((NW / 4 + 255) / 256), 256, 0, stream>>>(wq, wqh, NW / 4);
  cvt_f32_f16<<<(unsigned)((NW / 4 + 255) / 256), 256, 0, stream>>>(wk, wkh, NW / 4);
  cvt_f32_f16<<<(unsigned)((NW / 4 + 255) / 256), 256, 0, stream>>>(wv, wvh, NW / 4);
  cvt_f32_f16<<<(unsigned)((NW / 4 + 255) / 256), 256, 0, stream>>>(wu, wuh, NW / 4);
  cvt_f32_f16<<<(unsigned)((NW / 4 + 255) / 256), 256, 0, stream>>>(wo, woh, NW / 4);

  // 2) projections (Q natural, K/V transposed per head, U with silu)
  dim3 gproj(EMB / 128, MTOT / 128);
  gemm_proj<0><<<gproj, 256, 0, stream>>>(xh, wqh, Qb, EMB);
  gemm_proj<1><<<gproj, 256, 0, stream>>>(xh, wkh, Ktb, EMB);
  gemm_proj<2><<<gproj, 256, 0, stream>>>(xh, wvh, Vtb, EMB);
  gemm_proj<3><<<gproj, 256, 0, stream>>>(xh, wuh, Ub, EMB);

  // 3) KVt[e][d] = sum_l V[l,e] K[l,d]
  gemm_kv<<<BATCH * HEADS, 256, 0, stream>>>(Vtb, Ktb, KVb);

  // 4) O = Q @ KV (per head)
  gemm_o<<<dim3(SEQ / 128, BATCH * HEADS), 256, 0, stream>>>(Qb, KVb, Ob);

  // 5) SRMSNorm * silu(U), in place
  srms_mul<<<(MTOT * HEADS) / 8, 256, 0, stream>>>(Ob, Ub);

  // 6) out = O @ wo^T (fp32)
  gemm_proj<4><<<gproj, 256, 0, stream>>>(Ob, woh, d_out, EMB);
}
